// GCNEncoder_61220463837357
// MI455X (gfx1250) — matrix operand order repaired and checked by reading, not compiled, not run on hardware
//
#include <hip/hip_runtime.h>
#include <hip/hip_bf16.h>

// GCN 2-layer encoder for MI455X (gfx1250, wave32, WMMA).
// out[d] = dis[d] * ( sum_{e:dst=d} hs[src_e] + hs[d] ) + b,  hs[i] = dis[i]*(X@W)[i]

typedef __attribute__((ext_vector_type(16))) __bf16 v16bf;
typedef __attribute__((ext_vector_type(8)))  float  v8f;

#define D 128
#define KC 4                 // K chunks of 32 (K = 128)
#define NT 8                 // N tiles of 16 (D = 128)
#define WPB 8                // waves per block (256 threads)

__global__ __launch_bounds__(256) void k_init_deg(int* deg, int n) {
  int i = blockIdx.x * 256 + threadIdx.x;
  if (i < n) deg[i] = 1;                       // self-loop
}

__global__ __launch_bounds__(256) void k_count_deg(const int* __restrict__ dst,
                                                   int* deg, int e) {
  int i = blockIdx.x * 256 + threadIdx.x;
  if (i < e) atomicAdd(&deg[dst[i]], 1);
}

__global__ __launch_bounds__(256) void k_dis(const int* __restrict__ deg,
                                             float* dis, int n) {
  int i = blockIdx.x * 256 + threadIdx.x;
  if (i < n) dis[i] = rsqrtf((float)deg[i]);
}

// Hs = rowscale(dis) * (X @ W), also written to Acc as the self-loop init.
// One wave computes a 16-row stripe across all 128 output columns.
// Safe to run in-place (X == Hs): each wave reads only its own 16 rows,
// completely, before writing them.
__global__ __launch_bounds__(256) void k_gemm_rowscale(
    const float* X, const float* __restrict__ W,
    const float* __restrict__ dis, float* Hs,
    float* __restrict__ Acc, int n_rows)
{
  // W pre-permuted to bf16 B-fragment layout: [kc][nt][lane][16 bf16] = 32 KB
  __shared__ __align__(32) __bf16 ldsB[KC * NT * 32 * 16];

  const int t = threadIdx.x;
  for (int idx = t; idx < KC * NT * 32 * 16; idx += 256) {
    const int i    = idx & 15;          // position within lane's 16 values
    const int lane = (idx >> 4) & 31;
    const int nt   = (idx >> 9) & 7;
    const int kc   = idx >> 12;
    // B (32x16, 16-bit), same K order as A: lanes 0-15 hold K=0..7 then 16..23,
    // lanes 16-31 hold K=8..15 then 24..31.
    const int k = kc * 32 + (i & 7) + ((i & 8) ? 16 : 0) + ((lane & 16) ? 8 : 0);
    const int n = nt * 16 + (lane & 15);
    ldsB[idx] = (__bf16)W[k * D + n];
  }
  __syncthreads();

  const int wave = t >> 5;
  const int lane = t & 31;
  const int row_base = (blockIdx.x * WPB + wave) * 16;
  if (row_base >= n_rows) return;

  v8f acc[NT] = {};

  // A (16x32, 16-bit) layout: lane<16 -> M=lane,   K = 0..7 then 16..23
  //                           lane>=16 -> M=lane-16, K = 8..15 then 24..31
  const int m  = row_base + (lane & 15);
  const int kA = (lane & 16) ? 8 : 0;
  const float* xrow = X + (size_t)m * D;

  #pragma unroll
  for (int kc = 0; kc < KC; ++kc) {
    float a[16];
    const float* xp = xrow + kc * 32 + kA;
    *(float4*)(&a[0])  = *(const float4*)(xp + 0);
    *(float4*)(&a[4])  = *(const float4*)(xp + 4);
    *(float4*)(&a[8])  = *(const float4*)(xp + 16);
    *(float4*)(&a[12]) = *(const float4*)(xp + 20);
    v16bf af;
    #pragma unroll
    for (int i = 0; i < 16; ++i) af[i] = (__bf16)a[i];

    #pragma unroll
    for (int nt = 0; nt < NT; ++nt) {
      const v16bf bf = *(const v16bf*)(&ldsB[((kc * NT + nt) * 32 + lane) * 16]);
      acc[nt] = __builtin_amdgcn_wmma_f32_16x16x32_bf16(
          /*neg_a=*/false, af, /*neg_b=*/false, bf,
          /*c_mod=*/(short)0, acc[nt], /*reuse_a=*/false, /*reuse_b=*/false);
    }
  }

  // C/D layout: lane L: N = (L&15)+16*nt; VGPR j: M = j + (L>=16 ? 8 : 0)
  const int mhalf = (lane & 16) ? 8 : 0;
  float dv[8];
  #pragma unroll
  for (int j = 0; j < 8; ++j) dv[j] = dis[row_base + mhalf + j];

  #pragma unroll
  for (int nt = 0; nt < NT; ++nt) {
    const int n = nt * 16 + (lane & 15);
    #pragma unroll
    for (int j = 0; j < 8; ++j) {
      const size_t off = (size_t)(row_base + mhalf + j) * D + n;
      const float v = acc[nt][j] * dv[j];
      Hs[off]  = v;   // gather source
      Acc[off] = v;   // accumulator init (self-loop term)
    }
  }
}

// One wave per edge: gather Hs[src] (512 B), scatter-add into Acc[dst]
// with hardware f32 atomics (L2-resident working set on MI455X's 192 MB L2).
__global__ __launch_bounds__(256) void k_scatter(
    const int* __restrict__ src, const int* __restrict__ dst,
    const float* __restrict__ Hs, float* __restrict__ Acc, int n_edges)
{
  const int wid  = (int)((blockIdx.x * 256u + threadIdx.x) >> 5);
  const int lane = threadIdx.x & 31;
  if (wid >= n_edges) return;
  const int s = src[wid];
  const int d = dst[wid];
  const float4 v = *(const float4*)(Hs + (size_t)s * D + lane * 4);
  float* p = Acc + (size_t)d * D + lane * 4;
  unsafeAtomicAdd(p + 0, v.x);
  unsafeAtomicAdd(p + 1, v.y);
  unsafeAtomicAdd(p + 2, v.z);
  unsafeAtomicAdd(p + 3, v.w);
}

template <bool RELU>
__global__ __launch_bounds__(256) void k_epilogue(
    const float* __restrict__ Acc, const float* __restrict__ dis,
    const float* __restrict__ bias, float* __restrict__ out, int n_rows)
{
  const int tid = blockIdx.x * 256 + threadIdx.x;
  if (tid >= n_rows * (D / 4)) return;
  const int row = tid >> 5;           // D/4 == 32 float4 per row
  const int c   = (tid & 31) * 4;
  const float dvv = dis[row];
  const float4 a = *(const float4*)(Acc + (size_t)row * D + c);
  const float4 b = *(const float4*)(bias + c);
  float4 o;
  o.x = fmaf(dvv, a.x, b.x);
  o.y = fmaf(dvv, a.y, b.y);
  o.z = fmaf(dvv, a.z, b.z);
  o.w = fmaf(dvv, a.w, b.w);
  if (RELU) {
    o.x = fmaxf(o.x, 0.0f); o.y = fmaxf(o.y, 0.0f);
    o.z = fmaxf(o.z, 0.0f); o.w = fmaxf(o.w, 0.0f);
  }
  *(float4*)(out + (size_t)row * D + c) = o;
}

extern "C" void kernel_launch(void* const* d_in, const int* in_sizes, int n_in,
                              void* d_out, int out_size, void* d_ws, size_t ws_size,
                              hipStream_t stream)
{
  const float* x    = (const float*)d_in[0];
  const int*   eidx = (const int*)  d_in[1];
  const float* W1   = (const float*)d_in[2];
  const float* b1   = (const float*)d_in[3];
  const float* W2   = (const float*)d_in[4];
  const float* b2   = (const float*)d_in[5];

  const int N = in_sizes[0] / D;     // 50000
  const int E = in_sizes[1] / 2;     // 800000
  const int* src = eidx;
  const int* dst = eidx + E;

  char* ws = (char*)d_ws;
  size_t off = 0;
  auto alloc = [&](size_t bytes) -> char* {
    char* p = ws + off;
    off = (off + bytes + 255) & ~(size_t)255;
    return p;
  };
  int*   deg = (int*)  alloc((size_t)N * sizeof(int));
  float* dis = (float*)alloc((size_t)N * sizeof(float));
  float* B0  = (float*)alloc((size_t)N * D * sizeof(float));  // hs / layer input
  float* B1  = (float*)alloc((size_t)N * D * sizeof(float));  // accumulator
  (void)ws_size; (void)n_in; (void)out_size;

  const int nb  = (N + 255) / 256;
  const int eb  = (E + 255) / 256;
  const int gtb = ((N + 15) / 16 + WPB - 1) / WPB;  // gemm blocks
  const int sb  = (E + 7) / 8;                      // scatter blocks (1 wave/edge)
  const int pb  = (N * (D / 4) + 255) / 256;        // epilogue blocks

  // degrees + symmetric-norm scale
  k_init_deg <<<nb, 256, 0, stream>>>(deg, N);
  k_count_deg<<<eb, 256, 0, stream>>>(dst, deg, E);
  k_dis      <<<nb, 256, 0, stream>>>(deg, dis, N);

  // layer 1
  k_gemm_rowscale<<<gtb, 256, 0, stream>>>(x, W1, dis, B0, B1, N);
  k_scatter      <<<sb,  256, 0, stream>>>(src, dst, B0, B1, E);
  k_epilogue<true><<<pb, 256, 0, stream>>>(B1, dis, b1, B0, N);

  // layer 2 (GEMM in-place on B0: per-wave rows read fully before write)
  k_gemm_rowscale<<<gtb, 256, 0, stream>>>(B0, W2, dis, B0, B1, N);
  k_scatter      <<<sb,  256, 0, stream>>>(src, dst, B0, B1, E);
  k_epilogue<false><<<pb, 256, 0, stream>>>(B1, dis, b2, (float*)d_out, N);
}
